// graphModel_39805756900005
// MI455X (gfx1250) — matrix operand order repaired and checked by reading, not compiled, not run on hardware
//
#include <hip/hip_runtime.h>
#include <hip/hip_bf16.h>

// ---------------------------------------------------------------------------
// 2-layer GCN:  h = tanh(GCNConv(x,W1,b1)); h = tanh(GCNConv(h,W2,b2));
// out = mean(h[roots])
// GEMMs run through v_wmma_f32_16x16x32_bf16; aggregation via float atomics.
// ---------------------------------------------------------------------------

#define D_FEAT 768
#define BM 128
#define BN 64
#define BK 32

typedef __bf16 v16bf __attribute__((ext_vector_type(16)));
typedef float  v8f   __attribute__((ext_vector_type(8)));

static __device__ __forceinline__ unsigned short f2bf(float f) {
  unsigned int u = __float_as_uint(f);
  unsigned int r = u + 0x7FFFu + ((u >> 16) & 1u);   // round-to-nearest-even
  return (unsigned short)(r >> 16);
}

// ---- degree / norm ---------------------------------------------------------

__global__ void k_deg_init(float* deg, int n) {
  int i = blockIdx.x * blockDim.x + threadIdx.x;
  if (i < n) deg[i] = 1.0f;                       // self loop
}

__global__ void k_deg_scatter(const long long* __restrict__ ei, float* deg, int e) {
  int i = blockIdx.x * blockDim.x + threadIdx.x;
  if (i < e) atomicAdd(&deg[(int)ei[(size_t)e + i]], 1.0f);  // dst row
}

__global__ void k_dinv(float* deg, int n) {
  int i = blockIdx.x * blockDim.x + threadIdx.x;
  if (i < n) deg[i] = rsqrtf(deg[i]);             // deg >= 1 always
}

__global__ void k_edge_norm(const long long* __restrict__ ei,
                            const float* __restrict__ dinv,
                            float* __restrict__ norm, int e) {
  int i = blockIdx.x * blockDim.x + threadIdx.x;
  if (i < e)
    norm[i] = dinv[(int)ei[i]] * dinv[(int)ei[(size_t)e + i]];
}

// ---- casts -----------------------------------------------------------------

__global__ void k_cast_x(const float* __restrict__ x,
                         unsigned short* __restrict__ a, int total4) {
  int i = blockIdx.x * blockDim.x + threadIdx.x;
  if (i >= total4) return;
  float4 v = *(const float4*)(x + (size_t)i * 4);
  ushort4 o;
  o.x = f2bf(v.x); o.y = f2bf(v.y); o.z = f2bf(v.z); o.w = f2bf(v.w);
  *(ushort4*)(a + (size_t)i * 4) = o;
}

__global__ void k_tanh_cast(const float* __restrict__ agg,
                            unsigned short* __restrict__ a, int total4) {
  int i = blockIdx.x * blockDim.x + threadIdx.x;
  if (i >= total4) return;
  float4 v = *(const float4*)(agg + (size_t)i * 4);
  ushort4 o;
  o.x = f2bf(tanhf(v.x)); o.y = f2bf(tanhf(v.y));
  o.z = f2bf(tanhf(v.z)); o.w = f2bf(tanhf(v.w));
  *(ushort4*)(a + (size_t)i * 4) = o;
}

// Wt[n][k] = W[k][n], bf16
__global__ void k_wconv(const float* __restrict__ w, unsigned short* __restrict__ wt) {
  int i = blockIdx.x * blockDim.x + threadIdx.x;
  if (i >= D_FEAT * D_FEAT) return;
  int n = i / D_FEAT, k = i % D_FEAT;
  wt[i] = f2bf(w[(size_t)k * D_FEAT + n]);
}

// ---- WMMA GEMM:  C[MxD] = A[MxD](bf16) @ W[DxD]  (B given transposed) -----

__global__ __launch_bounds__(256) void k_gemm_bf16(
    const unsigned short* __restrict__ A,   // [M x 768] bf16 row-major
    const unsigned short* __restrict__ Bt,  // [768 x 768] bf16, Bt[n][k]=W[k][n]
    float* __restrict__ C, int M) {
  __shared__ unsigned short sA[BM][BK + 8];   // 80B rows -> 16B aligned chunks
  __shared__ unsigned short sB[BN][BK + 8];

  const int tid   = threadIdx.x;
  const int lane  = tid & 31;
  const int wave  = tid >> 5;
  const int wm    = wave >> 1;        // 0..3  (M sub-tile)
  const int wn    = wave & 1;         // 0..1  (N sub-tile)
  const int l16   = lane & 15;
  const int lhalf = lane >> 4;        // K-half selector per ISA layouts
  const int blockM = blockIdx.x * BM;
  const int blockN = blockIdx.y * BN;

  v8f acc[2][2] = {};

  union Frag { v16bf v; uint4 q[2]; };

  for (int k0 = 0; k0 < D_FEAT; k0 += BK) {
    // ---- stage A tile: 128x32 bf16, 256 thr x 16 bf16 (2x b128) ----
    {
      int row = tid >> 1;
      int kp  = (tid & 1) * 16;
      int gr  = blockM + row;
      if (gr < M) {
        const unsigned short* src = A + (size_t)gr * D_FEAT + k0 + kp;
        *(uint4*)(&sA[row][kp])     = *(const uint4*)(src);
        *(uint4*)(&sA[row][kp + 8]) = *(const uint4*)(src + 8);
        if (k0 + BK < D_FEAT)
          __builtin_prefetch(src + BK, 0, 0);       // global_prefetch_b8
      } else {
        uint4 z = make_uint4(0, 0, 0, 0);
        *(uint4*)(&sA[row][kp])     = z;
        *(uint4*)(&sA[row][kp + 8]) = z;
      }
    }
    // ---- stage B tile: 64x32 bf16, 256 thr x 8 bf16 (1x b128) ----
    {
      int n  = tid >> 2;
      int kp = (tid & 3) * 8;
      *(uint4*)(&sB[n][kp]) =
          *(const uint4*)(Bt + (size_t)(blockN + n) * D_FEAT + k0 + kp);
    }
    __syncthreads();

    Frag af[2], bf[2];
#pragma unroll
    for (int i = 0; i < 2; ++i) {
      // A 16x32 (16-bit) per-lane: K = lhalf*8..+7  and  16+lhalf*8..+7
      int row = wm * 32 + i * 16 + l16;
      af[i].q[0] = *(const uint4*)(&sA[row][lhalf * 8]);
      af[i].q[1] = *(const uint4*)(&sA[row][16 + lhalf * 8]);
      // B 32x16 per-lane: column l16, K = lhalf*8..+7  and  16+lhalf*8..+7
      int n = wn * 32 + i * 16 + l16;
      bf[i].q[0] = *(const uint4*)(&sB[n][lhalf * 8]);
      bf[i].q[1] = *(const uint4*)(&sB[n][16 + lhalf * 8]);
    }
#pragma unroll
    for (int i = 0; i < 2; ++i)
#pragma unroll
      for (int j = 0; j < 2; ++j)
        acc[i][j] = __builtin_amdgcn_wmma_f32_16x16x32_bf16(
            false, af[i].v, false, bf[j].v, (short)0, acc[i][j], false, false);
    __syncthreads();
  }

  // ---- store: VGPR v -> row v (lanes 0-15) / row 8+v (lanes 16-31) ----
#pragma unroll
  for (int i = 0; i < 2; ++i) {
#pragma unroll
    for (int j = 0; j < 2; ++j) {
      int col = blockN + wn * 32 + j * 16 + l16;
#pragma unroll
      for (int v = 0; v < 8; ++v) {
        int row = blockM + wm * 32 + i * 16 + (lhalf ? 8 + v : v);
        if (row < M) C[(size_t)row * D_FEAT + col] = acc[i][j][v];
      }
    }
  }
}

// ---- aggregation -----------------------------------------------------------

// AGG[i][d] = b[d] + H[i][d] * dinv[i]^2   (bias + self loop)
__global__ void k_agg_init(const float* __restrict__ H,
                           const float* __restrict__ dinv,
                           const float* __restrict__ b,
                           float* __restrict__ agg, int n) {
  int i = blockIdx.x * blockDim.x + threadIdx.x;
  int total = n * (D_FEAT / 4);
  if (i >= total) return;
  int node = i / (D_FEAT / 4);
  int c    = (i % (D_FEAT / 4)) * 4;
  float di = dinv[node];
  float cf = di * di;
  float4 h = *(const float4*)(H + (size_t)node * D_FEAT + c);
  float4 o;
  o.x = b[c + 0] + h.x * cf;
  o.y = b[c + 1] + h.y * cf;
  o.z = b[c + 2] + h.z * cf;
  o.w = b[c + 3] + h.w * cf;
  *(float4*)(agg + (size_t)node * D_FEAT + c) = o;
}

// one block per edge, 192 threads x float4: AGG[dst] += H[src] * norm[e]
__global__ __launch_bounds__(192) void k_agg_scatter(
    const float* __restrict__ H, const long long* __restrict__ ei,
    const float* __restrict__ norm, float* __restrict__ agg, int e) {
  int edge = blockIdx.x;
  int s = (int)ei[edge];
  int d = (int)ei[(size_t)e + edge];
  float w = norm[edge];
  int c = threadIdx.x * 4;
  float4 h = *(const float4*)(H + (size_t)s * D_FEAT + c);
  float* out = agg + (size_t)d * D_FEAT + c;
  atomicAdd(out + 0, h.x * w);
  atomicAdd(out + 1, h.y * w);
  atomicAdd(out + 2, h.z * w);
  atomicAdd(out + 3, h.w * w);
}

// ---- root mean pool (tanh applied at roots == tanh everywhere, then pool) --

__global__ void k_root_pool(const float* __restrict__ agg,
                            const long long* __restrict__ roots,
                            float* __restrict__ out, int nroots) {
  int d = blockIdx.x * blockDim.x + threadIdx.x;
  if (d >= D_FEAT) return;
  float s = 0.0f;
  for (int r = 0; r < nroots; ++r)
    s += tanhf(agg[(size_t)roots[r] * D_FEAT + d]);
  out[d] = s / (float)nroots;
}

// ---------------------------------------------------------------------------

static inline size_t align256(size_t x) { return (x + 255) & ~(size_t)255; }

extern "C" void kernel_launch(void* const* d_in, const int* in_sizes, int n_in,
                              void* d_out, int out_size, void* d_ws, size_t ws_size,
                              hipStream_t stream) {
  const float*     x     = (const float*)d_in[0];
  const long long* ei    = (const long long*)d_in[1];
  const long long* roots = (const long long*)d_in[2];
  const float*     W1    = (const float*)d_in[3];
  const float*     b1    = (const float*)d_in[4];
  const float*     W2    = (const float*)d_in[5];
  const float*     b2    = (const float*)d_in[6];

  const int N  = in_sizes[0] / D_FEAT;
  const int E  = in_sizes[1] / 2;
  const int NR = in_sizes[2];

  char* ws = (char*)d_ws;
  size_t off = 0;
  float* dinv = (float*)(ws + off);          off += align256((size_t)N * 4);
  float* norm = (float*)(ws + off);          off += align256((size_t)E * 4);
  unsigned short* Abf = (unsigned short*)(ws + off); off += align256((size_t)N * D_FEAT * 2);
  unsigned short* Wt  = (unsigned short*)(ws + off); off += align256((size_t)D_FEAT * D_FEAT * 2);
  float* H   = (float*)(ws + off);           off += align256((size_t)N * D_FEAT * 4);
  float* AGG = (float*)(ws + off);           off += align256((size_t)N * D_FEAT * 4);
  (void)ws_size;

  const int totalND4 = N * (D_FEAT / 4);
  dim3 gemmGrid((N + BM - 1) / BM, D_FEAT / BN);

  // degree / norm (shared by both layers)
  k_deg_init   <<<(N + 255) / 256, 256, 0, stream>>>(dinv, N);
  k_deg_scatter<<<(E + 255) / 256, 256, 0, stream>>>(ei, dinv, E);
  k_dinv       <<<(N + 255) / 256, 256, 0, stream>>>(dinv, N);
  k_edge_norm  <<<(E + 255) / 256, 256, 0, stream>>>(ei, dinv, norm, E);

  // ---- layer 1 ----
  k_cast_x <<<(totalND4 + 255) / 256, 256, 0, stream>>>(x, Abf, totalND4);
  k_wconv  <<<(D_FEAT * D_FEAT + 255) / 256, 256, 0, stream>>>(W1, Wt);
  k_gemm_bf16<<<gemmGrid, 256, 0, stream>>>(Abf, Wt, H, N);
  k_agg_init <<<(totalND4 + 255) / 256, 256, 0, stream>>>(H, dinv, b1, AGG, N);
  k_agg_scatter<<<E, 192, 0, stream>>>(H, ei, norm, AGG, E);

  // ---- layer 2 ----
  k_tanh_cast<<<(totalND4 + 255) / 256, 256, 0, stream>>>(AGG, Abf, totalND4);
  k_wconv  <<<(D_FEAT * D_FEAT + 255) / 256, 256, 0, stream>>>(W2, Wt);
  k_gemm_bf16<<<gemmGrid, 256, 0, stream>>>(Abf, Wt, H, N);
  k_agg_init <<<(totalND4 + 255) / 256, 256, 0, stream>>>(H, dinv, b2, AGG, N);
  k_agg_scatter<<<E, 192, 0, stream>>>(H, ei, norm, AGG, E);

  // ---- pool ----
  k_root_pool<<<(D_FEAT + 255) / 256, 256, 0, stream>>>(AGG, roots, (float*)d_out, NR);
}
